// CausalSelfAttention_70188355551827
// MI455X (gfx1250) — matrix operand order repaired and checked by reading, not compiled, not run on hardware
//
#include <hip/hip_runtime.h>

// ---------- CDNA5 WMMA types ----------
typedef __attribute__((ext_vector_type(16))) __bf16 v16bf;
typedef __attribute__((ext_vector_type(8)))  __bf16 v8bf;
typedef __attribute__((ext_vector_type(2)))  __bf16 v2bf;
typedef __attribute__((ext_vector_type(8)))  float  v8f;

// Native bf16 conversions: v2bf init lowers to a single v_cvt_pk_bf16_f32.
static __device__ __forceinline__ unsigned short bf16_1(float x) {
    return __builtin_bit_cast(unsigned short, (__bf16)x);
}
static __device__ __forceinline__ unsigned int pack_bf16x2(float x, float y) {
    v2bf v = { (__bf16)x, (__bf16)y };
    return __builtin_bit_cast(unsigned int, v);
}

// A-fragment (16x32 bf16, MxK): lanes 0-15 -> M=lane, K=0..7 & 16..23;
// lanes 16-31 -> M=lane-16, K=8..15 & 24..31.  p = row start in LDS.
static __device__ __forceinline__ v16bf load_frag_a(const unsigned short* p, int lane) {
    int kb = (lane < 16) ? 0 : 8;
    v8bf lo = *(const v8bf*)(p + kb);
    v8bf hi = *(const v8bf*)(p + kb + 16);
    return __builtin_shufflevector(lo, hi, 0,1,2,3,4,5,6,7,8,9,10,11,12,13,14,15);
}

// B-fragment (32x16 bf16, KxN): lanes 0-15 -> N=lane, K=0..7 & 16..23;
// lanes 16-31 -> N=lane-16, K=8..15 & 24..31.  p = transposed n-row (contiguous K).
static __device__ __forceinline__ v16bf load_frag_b(const unsigned short* p, int lane) {
    int kb = (lane < 16) ? 0 : 8;
    v8bf lo = *(const v8bf*)(p + kb);
    v8bf hi = *(const v8bf*)(p + kb + 16);
    return __builtin_shufflevector(lo, hi, 0,1,2,3,4,5,6,7,8,9,10,11,12,13,14,15);
}

// =====================================================================
// GEMM with bias: C[M,N] = A[M,K] @ B[K,N] + bias[N]
// Block tile 128x128, K-step 32, 256 threads = 8 waves, double-buffered LDS.
// Wave w: rows w*16..w*16+15, all 128 cols -> 8 WMMAs per K-step.
// =====================================================================
#define GLDS 40           // LDS row stride (elements); 80B keeps 16B k-offset alignment
#define GBUF (128 * GLDS) // one buffer: 128 rows

static __device__ __forceinline__ void gemm_load_tiles(
    const float* __restrict__ A, const float* __restrict__ B,
    int N, int K, int mBase, int nBase, int kt, int tid,
    float4 aS[4], float4 bS[4]) {
    #pragma unroll
    for (int it = 0; it < 4; ++it) {
        int i   = tid + it * 256;
        int row = i >> 3;
        int col = (i & 7) * 4;
        aS[it] = *(const float4*)(A + (size_t)(mBase + row) * K + kt + col);
    }
    int k0 = (tid >> 5) * 4;        // 8 k-groups of 4
    int n0 = (tid & 31) * 4;        // 32 n-groups of 4 (coalesced across lanes)
    #pragma unroll
    for (int j = 0; j < 4; ++j)
        bS[j] = *(const float4*)(B + (size_t)(kt + k0 + j) * N + nBase + n0);
}

static __device__ __forceinline__ void gemm_store_tiles(
    unsigned short* __restrict__ Asb, unsigned short* __restrict__ Bsb,
    int tid, const float4 aS[4], const float4 bS[4]) {
    #pragma unroll
    for (int it = 0; it < 4; ++it) {
        int i   = tid + it * 256;
        int row = i >> 3;
        int col = (i & 7) * 4;
        *(uint2*)&Asb[row * GLDS + col] =
            make_uint2(pack_bf16x2(aS[it].x, aS[it].y), pack_bf16x2(aS[it].z, aS[it].w));
    }
    // 4x4 micro-transpose: B[k0..k0+3][n0..n0+3] -> Bs[n][k], packed b64 stores
    int k0 = (tid >> 5) * 4;
    int n0 = (tid & 31) * 4;
    *(uint2*)&Bsb[(n0 + 0) * GLDS + k0] =
        make_uint2(pack_bf16x2(bS[0].x, bS[1].x), pack_bf16x2(bS[2].x, bS[3].x));
    *(uint2*)&Bsb[(n0 + 1) * GLDS + k0] =
        make_uint2(pack_bf16x2(bS[0].y, bS[1].y), pack_bf16x2(bS[2].y, bS[3].y));
    *(uint2*)&Bsb[(n0 + 2) * GLDS + k0] =
        make_uint2(pack_bf16x2(bS[0].z, bS[1].z), pack_bf16x2(bS[2].z, bS[3].z));
    *(uint2*)&Bsb[(n0 + 3) * GLDS + k0] =
        make_uint2(pack_bf16x2(bS[0].w, bS[1].w), pack_bf16x2(bS[2].w, bS[3].w));
}

__global__ __launch_bounds__(256)
void wmma_gemm_bias_kernel(const float* __restrict__ A, const float* __restrict__ B,
                           const float* __restrict__ bias, float* __restrict__ C,
                           int M, int N, int K) {
    __shared__ __align__(16) unsigned short As[2 * GBUF];
    __shared__ __align__(16) unsigned short Bs[2 * GBUF];
    const int tid  = threadIdx.x;
    const int wave = tid >> 5;
    const int lane = tid & 31;
    const int lh   = lane & 15;
    const int mBase = blockIdx.y * 128;
    const int nBase = blockIdx.x * 128;

    const v8f vzero = {};
    v8f acc[8] = {vzero, vzero, vzero, vzero, vzero, vzero, vzero, vzero};

    float4 aS[4], bS[4];
    gemm_load_tiles(A, B, N, K, mBase, nBase, 0, tid, aS, bS);
    gemm_store_tiles(As, Bs, tid, aS, bS);
    __syncthreads();

    const int nk = K >> 5;
    for (int ki = 0; ki < nk; ++ki) {
        const unsigned short* Ac = As + (ki & 1) * GBUF;
        const unsigned short* Bc = Bs + (ki & 1) * GBUF;
        const bool more = (ki + 1 < nk);

        // issue next tile's global loads; they overlap the WMMA chain below
        if (more) gemm_load_tiles(A, B, N, K, mBase, nBase, (ki + 1) * 32, tid, aS, bS);

        // pre-load all fragments, then drain the WMMA chain
        v16bf a = load_frag_a(Ac + (wave * 16 + lh) * GLDS, lane);
        v16bf bfr[8];
        #pragma unroll
        for (int t = 0; t < 8; ++t)
            bfr[t] = load_frag_b(Bc + (t * 16 + lh) * GLDS, lane);
        #pragma unroll
        for (int t = 0; t < 8; ++t)
            acc[t] = __builtin_amdgcn_wmma_f32_16x16x32_bf16(
                false, a, false, bfr[t], (short)0, acc[t], false, false);

        if (more) {
            gemm_store_tiles(As + ((ki + 1) & 1) * GBUF, Bs + ((ki + 1) & 1) * GBUF,
                             tid, aS, bS);
            __syncthreads();   // single barrier per K-step (ping-pong buffers)
        }
    }

    // epilogue: lane holds col = t*16+lh, vgpr r -> row r (+8 for hi lane half)
    #pragma unroll
    for (int t = 0; t < 8; ++t) {
        int col = nBase + t * 16 + lh;
        float bv = bias[col];
        int rbase = mBase + wave * 16 + ((lane < 16) ? 0 : 8);
        #pragma unroll
        for (int r = 0; r < 8; ++r) {
            C[(size_t)(rbase + r) * N + col] = acc[t][r] + bv;
        }
    }
}

// =====================================================================
// Flash attention: qkv[B*S][3D] fp32 -> out[B*S][D] fp32
// Block: (b,h, 128 q rows), 8 waves; wave owns 16 q rows.
// Per 64-key block: S = Q@K^T (WMMA), mask, online softmax, O += P@V (WMMA).
// =====================================================================
#define ALDS 72  // 144B row stride, keeps 16B alignment of k-offsets

__global__ __launch_bounds__(256)
void flash_attn_kernel(const float* __restrict__ qkv, const int* __restrict__ mask,
                       float* __restrict__ out) {
    constexpr int Sq = 2048, Dm = 1024;
    constexpr float qscale = 0.03125f;  // 1/sqrt(1024) (ref scales by 1/sqrt(d_model))
    __shared__ __align__(16) unsigned short Qs[128 * ALDS];  // [q][d] bf16 (pre-scaled)
    __shared__ __align__(16) unsigned short Ks[64 * ALDS];   // [k][d] bf16
    __shared__ __align__(16) unsigned short Vt[64 * ALDS];   // [d][k] bf16 (transposed)
    __shared__ __align__(16) unsigned short Ps[128 * ALDS];  // [q][k] bf16 probs

    const int tid  = threadIdx.x;
    const int wave = tid >> 5;
    const int lane = tid & 31;
    const int lh   = lane & 15;
    const int b    = blockIdx.y >> 4;
    const int h    = blockIdx.y & 15;
    const int qBase = blockIdx.x * 128;
    const size_t rowStride = 3 * Dm;

    // ---- load + scale Q chunk (128 x 64) ----
    #pragma unroll
    for (int it = 0; it < 8; ++it) {
        int i   = tid + it * 256;
        int row = i >> 4;
        int col = (i & 15) * 4;
        const float4 v = *(const float4*)(qkv + (size_t)(b * Sq + qBase + row) * rowStride + h * 64 + col);
        *(uint2*)&Qs[row * ALDS + col] =
            make_uint2(pack_bf16x2(v.x * qscale, v.y * qscale),
                       pack_bf16x2(v.z * qscale, v.w * qscale));
    }

    const v8f vzero = {};
    v8f o[4] = {vzero, vzero, vzero, vzero};
    float m[8], l[8];
    #pragma unroll
    for (int r = 0; r < 8; ++r) { m[r] = -3.0e38f; l[r] = 0.0f; }

    // K/V loader decomposition: 4(d) x 4(key) micro-tile per thread
    const int d0   = (tid & 15) * 4;   // coalesced across adjacent lanes
    const int key0 = (tid >> 4) * 4;

    for (int kb0 = 0; kb0 < Sq; kb0 += 64) {
        // ---- load K block (row major) and V block (transposed), one pass ----
        {
            float4 kv[4], vv[4];
            #pragma unroll
            for (int j = 0; j < 4; ++j) {
                const float* base = qkv + (size_t)(b * Sq + kb0 + key0 + j) * rowStride + h * 64 + d0;
                kv[j] = *(const float4*)(base + Dm);
                vv[j] = *(const float4*)(base + 2 * Dm);
            }
            #pragma unroll
            for (int j = 0; j < 4; ++j)
                *(uint2*)&Ks[(key0 + j) * ALDS + d0] =
                    make_uint2(pack_bf16x2(kv[j].x, kv[j].y), pack_bf16x2(kv[j].z, kv[j].w));
            *(uint2*)&Vt[(d0 + 0) * ALDS + key0] =
                make_uint2(pack_bf16x2(vv[0].x, vv[1].x), pack_bf16x2(vv[2].x, vv[3].x));
            *(uint2*)&Vt[(d0 + 1) * ALDS + key0] =
                make_uint2(pack_bf16x2(vv[0].y, vv[1].y), pack_bf16x2(vv[2].y, vv[3].y));
            *(uint2*)&Vt[(d0 + 2) * ALDS + key0] =
                make_uint2(pack_bf16x2(vv[0].z, vv[1].z), pack_bf16x2(vv[2].z, vv[3].z));
            *(uint2*)&Vt[(d0 + 3) * ALDS + key0] =
                make_uint2(pack_bf16x2(vv[0].w, vv[1].w), pack_bf16x2(vv[2].w, vv[3].w));
        }
        __syncthreads();

        // ---- scores: S[16 x 64] = Qstrip @ K^T (K-dim = hd = 64 -> 2 wmma steps) ----
        v8f s[4] = {vzero, vzero, vzero, vzero};
        #pragma unroll
        for (int ds = 0; ds < 64; ds += 32) {
            v16bf a = load_frag_a(&Qs[(wave * 16 + lh) * ALDS + ds], lane);
            v16bf bfr[4];
            #pragma unroll
            for (int t = 0; t < 4; ++t)
                bfr[t] = load_frag_b(&Ks[(t * 16 + lh) * ALDS + ds], lane);
            #pragma unroll
            for (int t = 0; t < 4; ++t)
                s[t] = __builtin_amdgcn_wmma_f32_16x16x32_bf16(
                    false, a, false, bfr[t], (short)0, s[t], false, false);
        }

        // ---- padding mask (additive) ----
        float neg[4];
        #pragma unroll
        for (int t = 0; t < 4; ++t)
            neg[t] = (mask[b * Sq + kb0 + t * 16 + lh] == 0) ? -1.0e9f : 0.0f;

        // ---- online softmax per row slot; reductions over the 16-lane N group ----
        #pragma unroll
        for (int r = 0; r < 8; ++r) {
            float mx = -3.0e38f;
            #pragma unroll
            for (int t = 0; t < 4; ++t) { s[t][r] += neg[t]; mx = fmaxf(mx, s[t][r]); }
            #pragma unroll
            for (int off = 1; off < 16; off <<= 1)
                mx = fmaxf(mx, __shfl_xor(mx, off, 32));
            float mnew   = fmaxf(m[r], mx);
            float factor = __expf(m[r] - mnew);
            float rs = 0.0f;
            #pragma unroll
            for (int t = 0; t < 4; ++t) {
                float p = __expf(s[t][r] - mnew);
                s[t][r] = p;
                rs += p;
            }
            #pragma unroll
            for (int off = 1; off < 16; off <<= 1)
                rs += __shfl_xor(rs, off, 32);
            l[r] = l[r] * factor + rs;
            m[r] = mnew;
            #pragma unroll
            for (int t = 0; t < 4; ++t) o[t][r] *= factor;
        }

        // ---- stage P to LDS (C-layout -> A-fragment layout conversion) ----
        #pragma unroll
        for (int t = 0; t < 4; ++t) {
            #pragma unroll
            for (int r = 0; r < 8; ++r) {
                int qr = wave * 16 + ((lane < 16) ? r : 8 + r);
                Ps[qr * ALDS + t * 16 + lh] = bf16_1(s[t][r]);
            }
        }
        __syncthreads();

        // ---- O += P @ V  (K-dim = 64 keys -> 2 wmma steps) ----
        #pragma unroll
        for (int ks = 0; ks < 64; ks += 32) {
            v16bf a = load_frag_a(&Ps[(wave * 16 + lh) * ALDS + ks], lane);
            v16bf bfr[4];
            #pragma unroll
            for (int t = 0; t < 4; ++t)
                bfr[t] = load_frag_b(&Vt[(t * 16 + lh) * ALDS + ks], lane);
            #pragma unroll
            for (int t = 0; t < 4; ++t)
                o[t] = __builtin_amdgcn_wmma_f32_16x16x32_bf16(
                    false, a, false, bfr[t], (short)0, o[t], false, false);
        }
        __syncthreads();

        // prefetch next K block (gfx1250 global_prefetch_b8)
        if (kb0 + 64 < Sq)
            __builtin_prefetch(qkv + (size_t)(b * Sq + kb0 + 64 + (tid & 63)) * rowStride + Dm + h * 64, 0, 0);
    }

    // ---- epilogue: normalize and write [b][s][h*64+d] ----
    #pragma unroll
    for (int t = 0; t < 4; ++t) {
        #pragma unroll
        for (int r = 0; r < 8; ++r) {
            int row = qBase + wave * 16 + ((lane < 16) ? r : 8 + r);
            out[(size_t)(b * Sq + row) * Dm + h * 64 + t * 16 + lh] = o[t][r] / l[r];
        }
    }
}

// =====================================================================
// Launch: qkv-GEMM -> flash attention -> projection GEMM
// =====================================================================
extern "C" void kernel_launch(void* const* d_in, const int* in_sizes, int n_in,
                              void* d_out, int out_size, void* d_ws, size_t ws_size,
                              hipStream_t stream) {
    const float* x      = (const float*)d_in[0];  // [2,2048,1024]
    const int*   amask  = (const int*)d_in[1];    // [2,2048]
    const float* w_attn = (const float*)d_in[2];  // [1024,3072]
    const float* b_attn = (const float*)d_in[3];  // [3072]
    const float* w_proj = (const float*)d_in[4];  // [1024,1024]
    const float* b_proj = (const float*)d_in[5];  // [1024]
    float* out = (float*)d_out;                   // [2,2048,1024]

    const int M = 4096;   // B*S
    const int D = 1024;
    const int N3 = 3072;

    float* qkv     = (float*)d_ws;                  // 4096*3072 floats
    float* attnout = qkv + (size_t)M * N3;          // 4096*1024 floats

    // 1) qkv = x @ w_attn + b_attn
    wmma_gemm_bias_kernel<<<dim3(N3 / 128, M / 128), 256, 0, stream>>>(
        x, w_attn, b_attn, qkv, M, N3, D);

    // 2) attention (flash, per (b,h,128-row q chunk))
    flash_attn_kernel<<<dim3(2048 / 128, 2 * 16), 256, 0, stream>>>(
        qkv, amask, attnout);

    // 3) out = attnout @ w_proj + b_proj
    wmma_gemm_bias_kernel<<<dim3(D / 128, M / 128), 256, 0, stream>>>(
        attnout, w_proj, b_proj, out, M, D, D);
}
